// LSTMGraves_69148973465944
// MI455X (gfx1250) — compile-verified, matrix operand order checked by reading, not run on hardware
//
#include <hip/hip_runtime.h>

#define DI __device__ __forceinline__

typedef __bf16 bf16_t;
typedef __bf16 v16bf __attribute__((ext_vector_type(16)));
typedef float  v8f   __attribute__((ext_vector_type(8)));

constexpr int B    = 64;
constexpr int T    = 1024;
constexpr int H    = 400;
constexpr int HP   = 448;          // K padded to 14*32 (7 per K-half)
constexpr int KT   = HP / 32;      // 14 wmma steps total
constexpr int KH   = KT / 2;       // 7 per wave (split-K)
constexpr int G4   = 1600;         // 4*H
constexpr int NW   = 100;          // persistent workgroups (25 col tiles x 4 row tiles)
constexpr int OUTC = 121;
constexpr int ROWB = HP * 2;       // bytes per padded row (896)
constexpr int SEG16 = ROWB / 16;   // 16B transfers per row (56)
constexpr int FCK  = 3 * HP;       // fc weight row length (1344)

constexpr size_t OUT_Y    = (size_t)B * T * OUTC;
constexpr size_t HOUT_OFF = OUT_Y;
constexpr size_t COUT_OFF = OUT_Y + (size_t)3 * B * H;

// ---- workspace layout (bytes) ----
constexpr size_t WMAT_B   = (size_t)G4 * HP * 2;            // 1,433,600
constexpr size_t OFF_SYNC = 0;                              // 2 x u32
constexpr size_t OFF_WHH  = 256;                            // 3 recurrent weight mats
constexpr size_t OFF_WIH  = OFF_WHH + 3 * WMAT_B;           // 2 skip-input weight mats
constexpr size_t OFF_FCW  = OFF_WIH + 2 * WMAT_B;           // [128][1344] bf16
constexpr size_t FCW_B    = (size_t)128 * FCK * 2;          // 344,064 (256-mult)
constexpr size_t OFF_H    = OFF_FCW + FCW_B;                // [64][448] bf16
constexpr size_t OFF_OUTS = OFF_H + (size_t)B * HP * 2;     // 3 x [T][64][448] bf16
constexpr size_t OUTS_E   = (size_t)T * B * HP;             // elements per layer

DI bf16_t f2bf(float f) {
  unsigned u = __builtin_bit_cast(unsigned, f);
  unsigned r = u + 0x7FFFu + ((u >> 16) & 1u);
  unsigned short h = (unsigned short)(r >> 16);
  return __builtin_bit_cast(bf16_t, h);
}
DI float sigm(float x)   { return 1.0f / (1.0f + __expf(-x)); }
DI float tanh_f(float x) { return 1.0f - 2.0f / (__expf(2.0f * x) + 1.0f); }

// CDNA5 async global->LDS copy (ASYNCcnt-tracked), per ISA 15.18.3
DI void async_ld_b128(unsigned lds_off, const void* gaddr) {
  asm volatile("global_load_async_to_lds_b128 %0, %1, off"
               :: "v"(lds_off), "v"(gaddr) : "memory");
}
DI void wait_async0() { asm volatile("s_wait_asynccnt 0x0" ::: "memory"); }

// ---- prep: convert fp32 weights to K-padded bf16 [rows][448] ----
__global__ void conv_pad_bf16(const float* __restrict__ src, bf16_t* __restrict__ dst,
                              int rows, int cols, int sstride, int coff) {
  int total = rows * HP;
  for (int i = blockIdx.x * blockDim.x + threadIdx.x; i < total;
       i += gridDim.x * blockDim.x) {
    int r = i / HP, c = i - r * HP;
    float v = (c < cols) ? src[(size_t)r * sstride + coff + c] : 0.0f;
    dst[i] = f2bf(v);
  }
}

// ---- prep: w_fc [121][1200] -> [128][3*448] bf16 (per-segment K padding) ----
__global__ void conv_fc(const float* __restrict__ wfc, bf16_t* __restrict__ dst) {
  int total = 128 * FCK;
  for (int i = blockIdx.x * blockDim.x + threadIdx.x; i < total;
       i += gridDim.x * blockDim.x) {
    int r = i / FCK, c = i - r * FCK;
    int seg = c / HP, k = c - seg * HP;
    float v = (r < OUTC && k < H) ? wfc[(size_t)r * 1200 + seg * H + k] : 0.0f;
    dst[i] = f2bf(v);
  }
}

// ---- prep: zero h buffer, zero pad columns of the 3 output tensors, reset barrier ----
__global__ void zero_init(bf16_t* __restrict__ hcur, bf16_t* __restrict__ outs,
                          unsigned* __restrict__ sync) {
  size_t tid = (size_t)blockIdx.x * blockDim.x + threadIdx.x;
  size_t stride = (size_t)gridDim.x * blockDim.x;
  if (tid == 0) sync[0] = 0u;                       // count=0; gen left monotonic
  bf16_t z = __builtin_bit_cast(bf16_t, (unsigned short)0);
  for (size_t i = tid; i < (size_t)B * HP; i += stride) hcur[i] = z;
  size_t pN = (size_t)3 * T * B;                    // one 48-wide pad stripe each
  for (size_t i = tid; i < pN; i += stride) {
    size_t l = i / ((size_t)T * B), rem = i - l * (size_t)T * B;
    bf16_t* p = outs + l * OUTS_E + rem * HP + H;
#pragma unroll
    for (int k = 0; k < 48; ++k) p[k] = z;
  }
}

// ---- grid-wide sense barrier (agent scope) ----
DI void grid_sync(unsigned* cnt, unsigned* gen) {
  __builtin_amdgcn_fence(__ATOMIC_RELEASE, "agent");
  __syncthreads();
  if (threadIdx.x == 0) {
    unsigned g = __hip_atomic_load(gen, __ATOMIC_RELAXED, __HIP_MEMORY_SCOPE_AGENT);
    unsigned a = __hip_atomic_fetch_add(cnt, 1u, __ATOMIC_ACQ_REL, __HIP_MEMORY_SCOPE_AGENT);
    if (a == (unsigned)(NW - 1)) {
      __hip_atomic_store(cnt, 0u, __ATOMIC_RELAXED, __HIP_MEMORY_SCOPE_AGENT);
      __hip_atomic_fetch_add(gen, 1u, __ATOMIC_RELEASE, __HIP_MEMORY_SCOPE_AGENT);
    } else {
      while (__hip_atomic_load(gen, __ATOMIC_ACQUIRE, __HIP_MEMORY_SCOPE_AGENT) == g)
        __builtin_amdgcn_s_sleep(1);
    }
  }
  __syncthreads();
  __builtin_amdgcn_fence(__ATOMIC_ACQUIRE, "agent");
}

// ---- persistent recurrent kernel: 100 WGs x 256 threads (8 waves, split-K) ----
__global__ __launch_bounds__(256)
void lstm_persistent(const float* __restrict__ x, const float* __restrict__ h0,
                     const float* __restrict__ c0,
                     const float* __restrict__ wih0, const float* __restrict__ wih1,
                     const float* __restrict__ wih2,
                     const float* __restrict__ bih0, const float* __restrict__ bhh0,
                     const float* __restrict__ bih1, const float* __restrict__ bhh1,
                     const float* __restrict__ bih2, const float* __restrict__ bhh2,
                     const bf16_t* __restrict__ whh_bf, const bf16_t* __restrict__ wihh_bf,
                     bf16_t* __restrict__ hcur, bf16_t* __restrict__ outs,
                     float* __restrict__ dout, unsigned* cnt, unsigned* gen) {
  const int wg   = blockIdx.x;
  const int colt = wg % 25, mt = wg / 25;
  const int b0 = mt * 16, j0 = colt * 16;
  const int lane  = threadIdx.x & 31;
  const int wave  = threadIdx.x >> 5;       // 0..7
  const int wgate = wave & 3;               // gate (i,f,g,o)
  const int khalf = wave >> 2;              // K half (0: kt 0..6, 1: kt 7..13)
  const int lm = lane & 15;                 // tile row (A) / tile col (B,C)
  const int lk = (lane >> 4) * 16;          // K sub-base within 32-chunk
  const int mb = (lane >> 4) << 3;          // C row base per half-wave
  const int kbase = khalf * KH * 32;        // element offset of this wave's K half

  __shared__ bf16_t ldsA[16 * HP];          // h_{t-1} tile (rows b0..b0+15)
  __shared__ bf16_t ldsP[16 * HP];          // out_{l-1,t} tile
  __shared__ float glds[2][4][16][16];      // split-K partial gate tiles
  __shared__ float cs[16][16];
  __shared__ float xs[16][3];

  const unsigned ldsA_off = (unsigned)(uintptr_t)&ldsA[0];
  const unsigned ldsP_off = (unsigned)(uintptr_t)&ldsP[0];

  for (int l = 0; l < 3; ++l) {
    const float* wih = (l == 0) ? wih0 : (l == 1 ? wih1 : wih2);
    const float* bih = (l == 0) ? bih0 : (l == 1 ? bih1 : bih2);
    const float* bhh = (l == 0) ? bhh0 : (l == 1 ? bhh1 : bhh2);
    const int wst = (l == 0) ? 3 : 403;
    const bf16_t* whh   = whh_bf + (size_t)l * G4 * HP;
    const bf16_t* wihh  = l ? (wihh_bf + (size_t)(l - 1) * G4 * HP) : nullptr;
    const bf16_t* aprev = l ? (outs + (size_t)(l - 1) * OUTS_E) : nullptr;
    bf16_t* outl = outs + (size_t)l * OUTS_E;

    // init owned slice of h (bf16) and resident cell state (LDS)
    {
      int m = threadIdx.x >> 4, j = threadIdx.x & 15;
      int b = b0 + m, jg = j0 + j;
      size_t src = (size_t)l * B * H + (size_t)b * H + jg;
      hcur[(size_t)b * HP + jg] = f2bf(h0[src]);
      cs[m][j] = c0[src];
    }
    grid_sync(cnt, gen);

    // per-lane loop invariants: gate column n, bias, x-projection weights
    const int n = wgate * H + j0 + lm;
    const float bias = bih[n] + bhh[n];
    const float wx0 = wih[(size_t)n * wst + 0];
    const float wx1 = wih[(size_t)n * wst + 1];
    const float wx2 = wih[(size_t)n * wst + 2];
    const bf16_t* bptr1 = whh + (size_t)n * HP + kbase + lk;
    const bf16_t* bptr2 = l ? (wihh + (size_t)n * HP + kbase + lk) : nullptr;
    const bf16_t* afragA = &ldsA[lm * HP + kbase + lk];
    const bf16_t* afragP = &ldsP[lm * HP + kbase + lk];

    // hoist this wave's B fragments into registers for the whole layer
    v16bf wb1[KH];
#pragma unroll
    for (int kt = 0; kt < KH; ++kt) wb1[kt] = *(const v16bf*)(bptr1 + kt * 32);
    v16bf wb2[KH];
    if (l) {
#pragma unroll
      for (int kt = 0; kt < KH; ++kt) wb2[kt] = *(const v16bf*)(bptr2 + kt * 32);
    }

    for (int t = 0; t < T; ++t) {
      // cooperative async copy of A tiles into LDS (overlaps with VALU below)
      {
        const char* hsrc = (const char*)(hcur + (size_t)b0 * HP);
        for (int i = threadIdx.x; i < 16 * SEG16; i += 256) {
          int r = i / SEG16, c16 = i - (i / SEG16) * SEG16;
          async_ld_b128(ldsA_off + r * ROWB + c16 * 16,
                        hsrc + (size_t)r * ROWB + c16 * 16);
        }
        if (l) {
          const char* psrc = (const char*)(aprev + ((size_t)t * B + b0) * HP);
          for (int i = threadIdx.x; i < 16 * SEG16; i += 256) {
            int r = i / SEG16, c16 = i - (i / SEG16) * SEG16;
            async_ld_b128(ldsP_off + r * ROWB + c16 * 16,
                          psrc + (size_t)r * ROWB + c16 * 16);
          }
        }
      }
      if (threadIdx.x < 48) {
        int r = threadIdx.x / 3, d = threadIdx.x - r * 3;
        xs[r][d] = x[((size_t)(b0 + r) * T + t) * 3 + d];
      }
      wait_async0();
      __syncthreads();        // LDS tiles + xs visible to all waves

      // C init: low-K waves carry bias + x_t @ w_ih_x^T; high-K waves start at 0
      v8f c;
      if (khalf == 0) {
#pragma unroll
        for (int v = 0; v < 8; ++v) {
          int m = v + mb;
          c[v] = bias + xs[m][0] * wx0 + xs[m][1] * wx1 + xs[m][2] * wx2;
        }
      } else {
#pragma unroll
        for (int v = 0; v < 8; ++v) c[v] = 0.0f;
      }

      // gates += h_{t-1} @ w_hh^T  (A from LDS, B register-resident)
#pragma unroll
      for (int kt = 0; kt < KH; ++kt) {
        v16bf a = *(const v16bf*)(afragA + kt * 32);
        c = __builtin_amdgcn_wmma_f32_16x16x32_bf16(false, a, false, wb1[kt],
                                                    (short)0, c, false, false);
      }
      // gates += out_{l-1,t} @ w_ih_h^T  (skip connection, layers 1..2)
      if (l) {
#pragma unroll
        for (int kt = 0; kt < KH; ++kt) {
          v16bf a = *(const v16bf*)(afragP + kt * 32);
          c = __builtin_amdgcn_wmma_f32_16x16x32_bf16(false, a, false, wb2[kt],
                                                      (short)0, c, false, false);
        }
      }
      // publish split-K partial gate tiles
#pragma unroll
      for (int v = 0; v < 8; ++v) glds[khalf][wgate][v + mb][lm] = c[v];
      __syncthreads();

      // elementwise LSTM cell update (cell state stays in LDS)
      {
        int m = threadIdx.x >> 4, j = threadIdx.x & 15;
        float iv = sigm(glds[0][0][m][j] + glds[1][0][m][j]);
        float fv = sigm(glds[0][1][m][j] + glds[1][1][m][j]);
        float gv = tanh_f(glds[0][2][m][j] + glds[1][2][m][j]);
        float ov = sigm(glds[0][3][m][j] + glds[1][3][m][j]);
        float cn = fv * cs[m][j] + iv * gv;
        cs[m][j] = cn;
        float hn = ov * tanh_f(cn);
        int b = b0 + m, jg = j0 + j;
        bf16_t hb = f2bf(hn);
        hcur[(size_t)b * HP + jg] = hb;
        outl[((size_t)t * B + b) * HP + jg] = hb;
        if (t == T - 1) {
          dout[HOUT_OFF + (size_t)l * B * H + (size_t)b * H + jg] = hn;
          dout[COUT_OFF + (size_t)l * B * H + (size_t)b * H + jg] = cn;
        }
      }
      grid_sync(cnt, gen);   // publish h_t to all workgroups
    }
  }
}

// ---- FC: concat(out0,out1,out2) [65536 x 1200] @ w_fc^T -> [B][T][121] ----
__global__ __launch_bounds__(128)
void fc_kernel(const bf16_t* __restrict__ outs, const bf16_t* __restrict__ fcw,
               const float* __restrict__ bfc, float* __restrict__ dout) {
  const int mtile = blockIdx.x;           // 0..4095
  const int t  = mtile >> 2;
  const int b0 = (mtile & 3) * 16;
  const int lane = threadIdx.x & 31;
  const int w  = threadIdx.x >> 5;        // 4 waves x 2 N-tiles = 128 cols
  const int lm = lane & 15;
  const int lk = (lane >> 4) * 16;
  const int mb = (lane >> 4) << 3;

  v8f acc[2];
#pragma unroll
  for (int nt = 0; nt < 2; ++nt)
#pragma unroll
    for (int v = 0; v < 8; ++v) acc[nt][v] = 0.0f;

#pragma unroll
  for (int seg = 0; seg < 3; ++seg) {
    const bf16_t* aseg = outs + (size_t)seg * OUTS_E + ((size_t)t * B + b0 + lm) * HP + lk;
#pragma unroll
    for (int kt = 0; kt < KT; ++kt) {
      v16bf a = *(const v16bf*)(aseg + kt * 32);
#pragma unroll
      for (int nt = 0; nt < 2; ++nt) {
        int n = (2 * w + nt) * 16 + lm;
        v16bf bm = *(const v16bf*)(fcw + (size_t)n * FCK + seg * HP + lk + kt * 32);
        acc[nt] = __builtin_amdgcn_wmma_f32_16x16x32_bf16(false, a, false, bm,
                                                          (short)0, acc[nt], false, false);
      }
    }
  }
#pragma unroll
  for (int nt = 0; nt < 2; ++nt) {
    int n = (2 * w + nt) * 16 + lm;
    if (n < OUTC) {
      float bb = bfc[n];
#pragma unroll
      for (int v = 0; v < 8; ++v)
        dout[((size_t)(b0 + v + mb) * T + t) * OUTC + n] = acc[nt][v] + bb;
    }
  }
}

extern "C" void kernel_launch(void* const* d_in, const int* in_sizes, int n_in,
                              void* d_out, int out_size, void* d_ws, size_t ws_size,
                              hipStream_t stream) {
  (void)in_sizes; (void)n_in; (void)out_size; (void)ws_size;
  const float* x    = (const float*)d_in[0];
  const float* h0   = (const float*)d_in[1];
  const float* c0   = (const float*)d_in[2];
  const float* wih0 = (const float*)d_in[3];
  const float* whh0 = (const float*)d_in[4];
  const float* bih0 = (const float*)d_in[5];
  const float* bhh0 = (const float*)d_in[6];
  const float* wih1 = (const float*)d_in[7];
  const float* whh1 = (const float*)d_in[8];
  const float* bih1 = (const float*)d_in[9];
  const float* bhh1 = (const float*)d_in[10];
  const float* wih2 = (const float*)d_in[11];
  const float* whh2 = (const float*)d_in[12];
  const float* bih2 = (const float*)d_in[13];
  const float* bhh2 = (const float*)d_in[14];
  const float* wfc  = (const float*)d_in[15];
  const float* bfc  = (const float*)d_in[16];

  char* ws = (char*)d_ws;
  unsigned* sync = (unsigned*)(ws + OFF_SYNC);
  bf16_t* whhb = (bf16_t*)(ws + OFF_WHH);
  bf16_t* wihb = (bf16_t*)(ws + OFF_WIH);
  bf16_t* fcwb = (bf16_t*)(ws + OFF_FCW);
  bf16_t* hcur = (bf16_t*)(ws + OFF_H);
  bf16_t* outs = (bf16_t*)(ws + OFF_OUTS);
  float* out = (float*)d_out;

  const int cgrid = (G4 * HP + 255) / 256;
  conv_pad_bf16<<<cgrid, 256, 0, stream>>>(whh0, whhb + 0 * (size_t)G4 * HP, G4, H, H, 0);
  conv_pad_bf16<<<cgrid, 256, 0, stream>>>(whh1, whhb + 1 * (size_t)G4 * HP, G4, H, H, 0);
  conv_pad_bf16<<<cgrid, 256, 0, stream>>>(whh2, whhb + 2 * (size_t)G4 * HP, G4, H, H, 0);
  conv_pad_bf16<<<cgrid, 256, 0, stream>>>(wih1, wihb + 0 * (size_t)G4 * HP, G4, H, 403, 3);
  conv_pad_bf16<<<cgrid, 256, 0, stream>>>(wih2, wihb + 1 * (size_t)G4 * HP, G4, H, 403, 3);
  conv_fc<<<(128 * FCK + 255) / 256, 256, 0, stream>>>(wfc, fcwb);
  zero_init<<<800, 256, 0, stream>>>(hcur, outs, sync);

  lstm_persistent<<<NW, 256, 0, stream>>>(x, h0, c0, wih0, wih1, wih2,
      bih0, bhh0, bih1, bhh1, bih2, bhh2, whhb, wihb, hcur, outs, out,
      sync, sync + 1);

  fc_kernel<<<T * 4, 128, 0, stream>>>(outs, fcwb, bfc, out);
}
